// DGIN_23759759081608
// MI455X (gfx1250) — matrix operand order repaired and checked by reading, not compiled, not run on hardware
//
#include <hip/hip_runtime.h>
#include <hip/hip_bf16.h>

// ---------------------------------------------------------------------------
// Types for CDNA5 WMMA (wave32): 16x16x32 bf16 -> f32
// ---------------------------------------------------------------------------
typedef __attribute__((ext_vector_type(16))) __bf16         v16bf;
typedef __attribute__((ext_vector_type(8)))  float          v8f;
typedef __attribute__((ext_vector_type(4)))  unsigned short v4u;
typedef __attribute__((ext_vector_type(8)))  unsigned short v8u;
typedef __attribute__((ext_vector_type(16))) unsigned short v16u;

// float -> bf16 (round to nearest even), as raw u16
__device__ __forceinline__ unsigned short f2bf(float f) {
  unsigned int u = __builtin_bit_cast(unsigned int, f);
  u += 0x7FFFu + ((u >> 16) & 1u);
  return (unsigned short)(u >> 16);
}

// ---------------------------------------------------------------------------
// WMMA GEMM: C[M,N] += A[M,K] * B[K,N], f32 in/out, bf16 compute, f32 accum.
// 256 threads = 8 waves; block does a 128-row strip, each wave a 16xN strip.
// Double-buffered LDS staging; K split over gridDim.y slices accumulated with
// native f32 atomics (C zeroed beforehand). NT = N/16 (4 or 8).
// ---------------------------------------------------------------------------
#define TM 128
#define TK 32
#define TKP 40              // padded LDS row stride in ushorts (80B, bank-safe)

template <int NT>
__global__ __launch_bounds__(256)
void k_gemm_wmma(const float* __restrict__ A, const float* __restrict__ B,
                 float* __restrict__ C, int M, int K)
{
  constexpr int N   = NT * 16;
  constexpr int NSH = (N == 64) ? 6 : 7;       // log2(N)
  constexpr int BPT = (TK / 4) * N / 256;      // B work items per thread (2|4)

  __shared__ __align__(16) unsigned short lds_a[2][TM * TKP];
  __shared__ __align__(16) unsigned short lds_b[2][128 * TKP];

  const int tid  = threadIdx.x;
  const int wave = tid >> 5;
  const int lane = tid & 31;
  const int half = lane >> 4;     // lane half
  const int l16  = lane & 15;
  const int m0   = blockIdx.x * TM;
  const int wrow = wave << 4;

  const int kPer = K / gridDim.y;
  const int kBeg = blockIdx.y * kPer;
  const int nk   = kPer / TK;

  // A staging: thread -> (row 0..127, k-half 0|16); 4x b128 loads, 2x b128 DS stores
  const int ar = tid >> 1;
  const int ak = (tid & 1) << 4;
  const float* agp = A + (size_t)(m0 + ar) * K + ak;

  float4 areg[4];
  float  breg[BPT][4];

  auto stage_global = [&](int k0) {
    const float4* ga = (const float4*)(agp + k0);
    #pragma unroll
    for (int i = 0; i < 4; ++i) areg[i] = ga[i];
    #pragma unroll
    for (int u = 0; u < BPT; ++u) {                 // B: 4 k of one column n
      int w  = tid + (u << 8);
      int n  = w & (N - 1);
      int kk = (w >> NSH) << 2;
      #pragma unroll
      for (int j = 0; j < 4; ++j)
        breg[u][j] = B[(size_t)(k0 + kk + j) * N + n];
    }
  };

  auto stage_lds = [&](int buf) {
    unsigned short* la = lds_a[buf];
    unsigned short* lb = lds_b[buf];
    const float* af = (const float*)areg;
    v8u lo, hi;
    #pragma unroll
    for (int j = 0; j < 8; ++j) { lo[j] = f2bf(af[j]); hi[j] = f2bf(af[8 + j]); }
    *(v8u*)&la[ar * TKP + ak]     = lo;
    *(v8u*)&la[ar * TKP + ak + 8] = hi;
    #pragma unroll
    for (int u = 0; u < BPT; ++u) {                 // col-major: lds_b[n*TKP+k]
      int w  = tid + (u << 8);
      int n  = w & (N - 1);
      int kk = (w >> NSH) << 2;
      v4u p;
      #pragma unroll
      for (int j = 0; j < 4; ++j) p[j] = f2bf(breg[u][j]);
      *(v4u*)&lb[n * TKP + kk] = p;
    }
  };

  v8f acc[NT] = {};

  stage_global(kBeg);
  stage_lds(0);
  __syncthreads();

  for (int it = 0; it < nk; ++it) {
    const int buf = it & 1;
    const int k0  = kBeg + it * TK;

    if (it + 1 < nk) stage_global(k0 + TK);   // fire next tile's global loads
    if (it + 2 < nk)                          // gfx1250 global_prefetch_b8
      __builtin_prefetch(agp + k0 + 2 * TK, 0, 1);

    // A fragment (16x32 bf16): element i -> k = (i>=8?16:0) + half*8 + (i&7)
    const v8u* pa = (const v8u*)&lds_a[buf][(wrow + l16) * TKP];
    v8u alo = pa[half];
    v8u ahi = pa[2 + half];
    v16u au;
    #pragma unroll
    for (int j = 0; j < 8; ++j) { au[j] = alo[j]; au[8 + j] = ahi[j]; }
    v16bf afr = __builtin_bit_cast(v16bf, au);

    #pragma unroll
    for (int nt = 0; nt < NT; ++nt) {         // B fragment: n=l16, k=(i>=8?16:0)+half*8+(i&7)
      const v8u* pb = (const v8u*)&lds_b[buf][((nt << 4) + l16) * TKP];
      v8u blo = pb[half];
      v8u bhi = pb[2 + half];
      v16u bu;
      #pragma unroll
      for (int j = 0; j < 8; ++j) { bu[j] = blo[j]; bu[8 + j] = bhi[j]; }
      v16bf bfr = __builtin_bit_cast(v16bf, bu);
      acc[nt] = __builtin_amdgcn_wmma_f32_16x16x32_bf16(
          false, afr, false, bfr, (short)0, acc[nt], false, false);
    }

    if (it + 1 < nk) stage_lds(buf ^ 1);      // wait loads, convert, fill other buf
    __syncthreads();
  }

  // epilogue: C/D layout lane l16 = col, VGPR r -> row r + 8*half
  #pragma unroll
  for (int nt = 0; nt < NT; ++nt) {
    #pragma unroll
    for (int r = 0; r < 8; ++r) {
      size_t idx = (size_t)(m0 + wrow + (half << 3) + r) * N + (nt << 4) + l16;
      unsafeAtomicAdd(&C[idx], acc[nt][r]);   // global_atomic_add_f32
    }
  }
}

// ---------------------------------------------------------------------------
// h0 = relu([eal | ef] @ W_init + b_init);  h = h0.   One thread per (e,j).
// ---------------------------------------------------------------------------
__global__ void k_init(const float* __restrict__ eal, const float* __restrict__ ef,
                       const float* __restrict__ Wi, const float* __restrict__ bi,
                       float* __restrict__ h0, float* __restrict__ h)
{
  int idx = blockIdx.x * blockDim.x + threadIdx.x;   // E*64
  int j = idx & 63, e = idx >> 6;
  const float* ea = eal + (size_t)e * 64;
  const float* ee = ef  + (size_t)e * 16;
  float s = bi[j];
  #pragma unroll 4
  for (int k = 0; k < 64; ++k) s += ea[k] * Wi[k * 64 + j];
  #pragma unroll 4
  for (int k = 0; k < 16; ++k) s += ee[k] * Wi[(64 + k) * 64 + j];
  s = fmaxf(s, 0.f);
  h0[idx] = s; h[idx] = s;
}

// ---------------------------------------------------------------------------
// Row layernorm in place: one wave per row (width 64 or 128), 8 rows / block.
// ---------------------------------------------------------------------------
__global__ void k_layernorm(float* __restrict__ X, const float* __restrict__ g,
                            const float* __restrict__ b, int R, int W)
{
  int row  = blockIdx.x * 8 + (threadIdx.x >> 5);
  int lane = threadIdx.x & 31;
  if (row >= R) return;
  float* x = X + (size_t)row * W;
  float v[4];
  int cnt = W >> 5;                       // 2 (W=64) or 4 (W=128)
  float s = 0.f;
  for (int i = 0; i < cnt; ++i) { v[i] = x[lane + 32 * i]; s += v[i]; }
  for (int o = 16; o > 0; o >>= 1) s += __shfl_xor(s, o, 32);
  float mu = s / (float)W;
  float q = 0.f;
  for (int i = 0; i < cnt; ++i) { float d = v[i] - mu; q += d * d; }
  for (int o = 16; o > 0; o >>= 1) q += __shfl_xor(q, o, 32);
  float rinv = rsqrtf(q / (float)W + 1e-6f);
  for (int i = 0; i < cnt; ++i) {
    int j = lane + 32 * i;
    x[j] = g[j] * (v[i] - mu) * rinv + b[j];
  }
}

// h = relu(h0 + m @ Wp)   (per-edge 64x64 weight mix)
__global__ void k_pass(const float* __restrict__ h0, const float* __restrict__ m,
                       const float* __restrict__ Wp, float* __restrict__ h)
{
  int idx = blockIdx.x * blockDim.x + threadIdx.x;   // E*64
  int j = idx & 63, e = idx >> 6;
  const float* mr = m + (size_t)e * 64;
  float s = h0[idx];
  #pragma unroll 4
  for (int k = 0; k < 64; ++k) s += mr[k] * Wp[k * 64 + j];
  h[idx] = fmaxf(s, 0.f);
}

// h0v = [nf | mv]; hv = h0v
__global__ void k_concat(const float* __restrict__ nf, const float* __restrict__ mv,
                         float* __restrict__ h0v, float* __restrict__ hv)
{
  int idx = blockIdx.x * blockDim.x + threadIdx.x;   // N*128
  int j = idx & 127, n = idx >> 7;
  float v = (j < 64) ? nf[(size_t)n * 64 + j] : mv[(size_t)n * 64 + (j - 64)];
  h0v[idx] = v; hv[idx] = v;
}

// hv = ((1+eps)*h0v + t) @ Wg + bg
__global__ void k_gin(const float* __restrict__ h0v, const float* __restrict__ t,
                      const float* __restrict__ eps, const float* __restrict__ Wg,
                      const float* __restrict__ bg, float* __restrict__ hv)
{
  int idx = blockIdx.x * blockDim.x + threadIdx.x;   // N*128
  int j = idx & 127, n = idx >> 7;
  const float* a = h0v + (size_t)n * 128;
  const float* c = t   + (size_t)n * 128;
  float s = bg[j];
  #pragma unroll 4
  for (int k = 0; k < 128; ++k)
    s += ((1.f + eps[k]) * a[k] + c[k]) * Wg[k * 128 + j];
  hv[idx] = s;
}

// partial column sums: each block sums 128 rows of [.,128]
__global__ void k_colsum(const float* __restrict__ x, float* __restrict__ part)
{
  __shared__ float sh[256];
  int j    = threadIdx.x & 127;
  int half = threadIdx.x >> 7;
  size_t base = (size_t)blockIdx.x * 128 * 128;
  float s = 0.f;
  for (int r = 0; r < 64; ++r)
    s += x[base + (size_t)(half * 64 + r) * 128 + j];
  sh[threadIdx.x] = s;
  __syncthreads();
  if (half == 0)
    part[(size_t)blockIdx.x * 128 + j] = sh[j] + sh[128 + j];
}

// final: sum partials, layernorm the single 128-wide row
__global__ void k_readout(const float* __restrict__ part, const float* __restrict__ g4,
                          const float* __restrict__ b4, float* __restrict__ out,
                          int nPart)
{
  __shared__ float sh[128];
  int j = threadIdx.x;                     // 128 threads
  float s = 0.f;
  for (int p = 0; p < nPart; ++p) s += part[(size_t)p * 128 + j];
  sh[j] = s;
  __syncthreads();
  float mu = 0.f;
  for (int i = 0; i < 128; ++i) mu += sh[i];
  mu *= (1.f / 128.f);
  float d = s - mu;
  __syncthreads();
  sh[j] = d * d;
  __syncthreads();
  float var = 0.f;
  for (int i = 0; i < 128; ++i) var += sh[i];
  var *= (1.f / 128.f);
  out[j] = g4[j] * d * rsqrtf(var + 1e-6f) + b4[j];
}

// ---------------------------------------------------------------------------
extern "C" void kernel_launch(void* const* d_in, const int* in_sizes, int n_in,
                              void* d_out, int out_size, void* d_ws, size_t ws_size,
                              hipStream_t stream)
{
  (void)in_sizes; (void)n_in; (void)out_size; (void)ws_size;
  const float* eal  = (const float*)d_in[0];
  const float* ef   = (const float*)d_in[1];
  const float* nf   = (const float*)d_in[2];
  const float* Ae   = (const float*)d_in[3];
  const float* Ane  = (const float*)d_in[4];
  const float* Aadj = (const float*)d_in[5];
  /* d_in[6] identity: unused by reference path */
  const float* Wi  = (const float*)d_in[7];
  const float* bi  = (const float*)d_in[8];
  const float* Wp  = (const float*)d_in[9];
  const float* Wg  = (const float*)d_in[10];
  const float* bg  = (const float*)d_in[11];
  const float* eps = (const float*)d_in[12];
  const float* g1  = (const float*)d_in[13];
  const float* b1  = (const float*)d_in[14];
  const float* g2  = (const float*)d_in[15];
  const float* b2  = (const float*)d_in[16];
  const float* g3  = (const float*)d_in[17];
  const float* b3  = (const float*)d_in[18];
  const float* g4  = (const float*)d_in[19];
  const float* b4  = (const float*)d_in[20];
  float* out = (float*)d_out;

  const int E = 8192, Nn = 4096, H = 64, G = 128;
  float* ws  = (float*)d_ws;
  float* h0  = ws;               // [E,64]
  float* h   = h0  + E * H;      // [E,64]
  float* m   = h   + E * H;      // [E,64]
  float* h0v = m   + E * H;      // [N,128]
  float* hv  = h0v + Nn * G;     // [N,128]
  float* tmp = hv  + Nn * G;     // [N,128] scratch (edge->node agg, partials)

  const int KS = 4;              // split-K slices for occupancy

  // h0 = relu([eal|ef] @ W_init + b); h = h0
  k_init<<<E * H / 256, 256, 0, stream>>>(eal, ef, Wi, bi, h0, h);

  // --- D-MPNN: 3 rounds of m = LN(A_e @ h); h = relu(h0 + m @ Wp[t]) ---
  for (int t = 0; t < 3; ++t) {
    hipMemsetAsync(m, 0, (size_t)E * H * sizeof(float), stream);
    k_gemm_wmma<4><<<dim3(E / TM, KS), 256, 0, stream>>>(Ae, h, m, E, E);
    k_layernorm<<<E / 8, 256, 0, stream>>>(m, g1, b1, E, 64);
    k_pass<<<E * H / 256, 256, 0, stream>>>(h0, m, Wp + t * 64 * 64, h);
  }

  // m_v = LN(A_ne @ h); h0_v = [nf | m_v]; hv = h0_v
  hipMemsetAsync(tmp, 0, (size_t)Nn * H * sizeof(float), stream);
  k_gemm_wmma<4><<<dim3(Nn / TM, KS), 256, 0, stream>>>(Ane, h, tmp, Nn, E);
  k_layernorm<<<Nn / 8, 256, 0, stream>>>(tmp, g2, b2, Nn, 64);
  k_concat<<<Nn * G / 256, 256, 0, stream>>>(nf, tmp, h0v, hv);

  // --- GIN: 3 rounds of hv = ((1+eps)*h0v + LN(A @ hv)) @ Wg[t] + bg[t] ---
  for (int t = 0; t < 3; ++t) {
    hipMemsetAsync(tmp, 0, (size_t)Nn * G * sizeof(float), stream);
    k_gemm_wmma<8><<<dim3(Nn / TM, KS), 256, 0, stream>>>(Aadj, hv, tmp, Nn, Nn);
    k_layernorm<<<Nn / 8, 256, 0, stream>>>(tmp, g3, b3, Nn, 128);
    k_gin<<<Nn * G / 256, 256, 0, stream>>>(h0v, tmp, eps,
                                            Wg + t * G * G, bg + t * G, hv);
  }

  // readout: column-sum then layernorm of the [1,128] row
  k_colsum<<<Nn / 128, 256, 0, stream>>>(hv, tmp);
  k_readout<<<1, 128, 0, stream>>>(tmp, g4, b4, out, Nn / 128);
}
